// AttentionSubsample_6253472383798
// MI455X (gfx1250) — compile-verified, matrix operand order checked by reading, not run on hardware
//
#include <hip/hip_runtime.h>

// ---------------- CDNA5 WMMA types ----------------
typedef __attribute__((ext_vector_type(16))) __bf16 v16bf;
typedef __attribute__((ext_vector_type(8)))  float  v8f;

union FragA { v16bf v; unsigned u[8]; };

// Problem constants (from reference setup)
#define NB     4
#define SLEN   4096
#define DMODEL 768
#define HNUM   12
#define HD     64
#define STRIDE 4
#define KLEN   1024          // SLEN / STRIDE
#define MQ     (NB*SLEN)     // 16384
#define MKV    (NB*KLEN)     // 4096

// f32 -> bf16 (round to nearest even), pure bit math (no header dependency)
__device__ __forceinline__ unsigned short f2bf(float f) {
  unsigned x = __float_as_uint(f);
  unsigned r = (x + 0x7fffu + ((x >> 16) & 1u)) >> 16;
  return (unsigned short)r;
}

// Starting K of 2-element pair p (0..7) for lane-half `half` in the 16-bit
// A/B WMMA fragment layout (ISA 7.12.2: VGPR0..3 hold K=kb..kb+7, VGPR4..7
// hold K=16+kb..16+kb+7, kb = 8*(lane/16)).
__device__ __forceinline__ int kpair(int half, int p) {
  int kb = half * 8;
  return (p < 4) ? (kb + 2 * p) : (16 + kb + 2 * (p - 4));
}

// One-instruction 16B global -> LDS copy (ASYNCcnt-tracked, no VGPR bounce).
__device__ __forceinline__ void async_g2l_b128(unsigned lds_off, const void* g) {
  asm volatile("global_load_async_to_lds_b128 %0, %1, off"
               :: "v"(lds_off), "v"((unsigned long long)(size_t)g) : "memory");
}
__device__ __forceinline__ void wait_async0() {
  asm volatile("s_wait_asynccnt 0" ::: "memory");
}

// ---------------- f32 -> bf16 convert ----------------
__global__ void cvt_f32_bf16_kernel(const float* __restrict__ in,
                                    unsigned short* __restrict__ out, int n) {
  int i = (blockIdx.x * 256 + threadIdx.x) * 4;
  if (i + 3 < n) {
    float4 f = *(const float4*)(in + i);
    uint2 o;
    o.x = (unsigned)f2bf(f.x) | ((unsigned)f2bf(f.y) << 16);
    o.y = (unsigned)f2bf(f.z) | ((unsigned)f2bf(f.w) << 16);
    *(uint2*)(out + i) = o;
  } else {
    for (; i < n; ++i) out[i] = f2bf(in[i]);
  }
}

// ---------------- generic bf16 WMMA GEMM ----------------
// C[m][n] = sum_k A[m*a_row_mul][k] * W[n][k]   (K = 768 fixed)
// Block tile 128(M) x 256(N), 8 waves as 2(M) x 4(N), wave tile 64x64.
// Double-buffered async LDS staging overlaps the tensor-DMA with the WMMAs.
// Cb != nullptr  -> bf16 output, else f32 output (+bias).
__global__ __launch_bounds__(256) void gemm_bf16_kernel(
    const unsigned short* __restrict__ A, int a_row_mul,
    const unsigned short* __restrict__ W,
    unsigned short* __restrict__ Cb, float* __restrict__ Cf,
    const float* __restrict__ bias, int ldc) {
  __shared__ unsigned short Al[2][128][40];   // 80B rows: 16B aligned, 20-dword
  __shared__ unsigned short Bl[2][256][40];

  const int t = threadIdx.x;
  const int lane = t & 31, wave = t >> 5;
  const int wm = wave & 1, wn = wave >> 1;       // 2(M) x 4(N) waves
  const int half = lane >> 4, l16 = lane & 15;
  const int bm = blockIdx.x, bn = blockIdx.y;

  auto stage = [&](int buf, int kk) {
#pragma unroll
    for (int i = 0; i < 2; ++i) {            // A: 512 16B-chunks / 256 thr
      int c = t + 256 * i;
      int row = c >> 2, col8 = (c & 3) * 8;
      long arow = (long)(bm * 128 + row) * a_row_mul;
      async_g2l_b128((unsigned)(size_t)&Al[buf][row][col8],
                     A + arow * 768 + kk + col8);
    }
#pragma unroll
    for (int i = 0; i < 4; ++i) {            // B: 1024 16B-chunks / 256 thr
      int c = t + 256 * i;
      int row = c >> 2, col8 = (c & 3) * 8;
      long wrow = (long)(bn * 256 + row);
      async_g2l_b128((unsigned)(size_t)&Bl[buf][row][col8],
                     W + wrow * 768 + kk + col8);
    }
  };

  v8f acc[4][4];
  v8f zero = {0.f,0.f,0.f,0.f,0.f,0.f,0.f,0.f};
#pragma unroll
  for (int s = 0; s < 4; ++s)
#pragma unroll
    for (int tt = 0; tt < 4; ++tt) acc[s][tt] = zero;

  stage(0, 0);
  wait_async0();
  __syncthreads();

  for (int it = 0; it < 24; ++it) {          // 768 / 32 K-steps
    const int cur = it & 1;
    if (it + 1 < 24) stage(cur ^ 1, (it + 1) * 32);   // overlap DMA w/ compute

    FragA a[4];
#pragma unroll
    for (int s = 0; s < 4; ++s) {
      int r = wm * 64 + s * 16 + l16;
#pragma unroll
      for (int p = 0; p < 8; ++p)
        a[s].u[p] = *(const unsigned*)&Al[cur][r][kpair(half, p)];
    }
#pragma unroll
    for (int tt = 0; tt < 4; ++tt) {
      FragA b;
      int r = wn * 64 + tt * 16 + l16;
#pragma unroll
      for (int p = 0; p < 8; ++p)
        b.u[p] = *(const unsigned*)&Bl[cur][r][kpair(half, p)];
#pragma unroll
      for (int s = 0; s < 4; ++s)
        acc[s][tt] = __builtin_amdgcn_wmma_f32_16x16x32_bf16(
            false, a[s].v, false, b.v, (short)0, acc[s][tt], false, false);
    }

    wait_async0();       // next tile landed
    __syncthreads();     // everyone done reading cur before it is overwritten
  }

  // ---- epilogue: single format branch, straight-line stores ----
  const long mbase = (long)bm * 128 + wm * 64;
  const long nbase = (long)bn * 256 + wn * 64 + l16;
  if (Cb) {
#pragma unroll
    for (int s = 0; s < 4; ++s)
#pragma unroll
      for (int v = 0; v < 8; ++v) {
        unsigned short* rowp = Cb + (mbase + s * 16 + v + 8 * half) * ldc + nbase;
#pragma unroll
        for (int tt = 0; tt < 4; ++tt)
          rowp[tt * 16] = f2bf(acc[s][tt][v]);
      }
  } else {
    float bv[4];
#pragma unroll
    for (int tt = 0; tt < 4; ++tt) bv[tt] = bias[nbase + tt * 16];
#pragma unroll
    for (int s = 0; s < 4; ++s)
#pragma unroll
      for (int v = 0; v < 8; ++v) {
        float* rowp = Cf + (mbase + s * 16 + v + 8 * half) * ldc + nbase;
#pragma unroll
        for (int tt = 0; tt < 4; ++tt)
          rowp[tt * 16] = acc[s][tt][v] + bv[tt];
      }
  }
}

// ---------------- fused flash attention ----------------
// grid: (SLEN/128, HNUM, NB), block 256. Wave w handles 16 queries.
// Q: [MQ][768] bf16 ; KV: [MKV][1536] bf16 (K cols 0..767, V cols 768..1535)
// O: [MQ][768] bf16
__global__ __launch_bounds__(256) void attn_kernel(
    const unsigned short* __restrict__ Q,
    const unsigned short* __restrict__ KV,
    unsigned short* __restrict__ O) {
  __shared__ unsigned short Kc[32][72];       // K chunk [k][d], 144B rows
  __shared__ unsigned short Vt[64][40];       // V chunk transposed [d][k], 80B rows
  __shared__ unsigned short Pw[8][16][40];    // per-wave P tile [m][k], 80B rows

  const int t = threadIdx.x, lane = t & 31, wave = t >> 5;
  const int half = lane >> 4, l16 = lane & 15;
  const int h = blockIdx.y, n = blockIdx.z;
  const long qrow0 = (long)n * SLEN + blockIdx.x * 128 + wave * 16;

  // Q A-fragments (held in registers across whole K loop); b128 global loads
  FragA qf[2];
  {
    const unsigned short* base = Q + (qrow0 + l16) * 768 + h * HD + half * 8;
#pragma unroll
    for (int dc = 0; dc < 2; ++dc) {
      uint4 lo = *(const uint4*)(base + dc * 32);
      uint4 hi = *(const uint4*)(base + dc * 32 + 16);
      qf[dc].u[0] = lo.x; qf[dc].u[1] = lo.y; qf[dc].u[2] = lo.z; qf[dc].u[3] = lo.w;
      qf[dc].u[4] = hi.x; qf[dc].u[5] = hi.y; qf[dc].u[6] = hi.z; qf[dc].u[7] = hi.w;
    }
  }

  v8f zero = {0.f,0.f,0.f,0.f,0.f,0.f,0.f,0.f};
  v8f o_[4];
#pragma unroll
  for (int tt = 0; tt < 4; ++tt) o_[tt] = zero;
  float mrun[8], lrun[8];
#pragma unroll
  for (int v = 0; v < 8; ++v) { mrun[v] = -1e30f; lrun[v] = 0.f; }

  const int kl = t >> 3;          // staging: 8 thr per k-row
  const int d8 = (t & 7) * 8;

  for (int c = 0; c < 32; ++c) {  // 32 chunks of 32 keys = KLEN
    __syncthreads();
    {   // stage K chunk (async direct-to-LDS) + transposed V chunk
      long kvrow = (long)n * KLEN + c * 32 + kl;
      async_g2l_b128((unsigned)(size_t)&Kc[kl][d8],
                     KV + kvrow * 1536 + h * HD + d8);
      uint4 vd = *(const uint4*)(KV + kvrow * 1536 + 768 + h * HD + d8);
      unsigned vv[4] = {vd.x, vd.y, vd.z, vd.w};
#pragma unroll
      for (int j = 0; j < 4; ++j) {
        Vt[d8 + 2 * j][kl]     = (unsigned short)(vv[j] & 0xffffu);
        Vt[d8 + 2 * j + 1][kl] = (unsigned short)(vv[j] >> 16);
      }
      if (c + 1 < 32)  // hint next chunk into cache (global_prefetch_b8)
        __builtin_prefetch(KV + (kvrow + 32) * 1536 + h * HD + d8, 0, 1);
    }
    wait_async0();
    __syncthreads();

    // scores: S = Q (16x64) . K^T (64x32)  -> two 16x16 D tiles
    v8f st[2];
#pragma unroll
    for (int kt = 0; kt < 2; ++kt) {
      FragA kb0, kb1;
      int r = kt * 16 + l16;
#pragma unroll
      for (int p = 0; p < 8; ++p) {
        kb0.u[p] = *(const unsigned*)&Kc[r][kpair(half, p)];
        kb1.u[p] = *(const unsigned*)&Kc[r][32 + kpair(half, p)];
      }
      st[kt] = __builtin_amdgcn_wmma_f32_16x16x32_bf16(
          false, qf[0].v, false, kb0.v, (short)0, zero, false, false);
      st[kt] = __builtin_amdgcn_wmma_f32_16x16x32_bf16(
          false, qf[1].v, false, kb1.v, (short)0, st[kt], false, false);
    }

    // online softmax: row (query) reductions across the 16-lane group
#pragma unroll
    for (int v = 0; v < 8; ++v) {
      float s0 = st[0][v] * 0.125f, s1 = st[1][v] * 0.125f;
      float r = fmaxf(s0, s1);
      r = fmaxf(r, __shfl_xor(r, 1, 32));
      r = fmaxf(r, __shfl_xor(r, 2, 32));
      r = fmaxf(r, __shfl_xor(r, 4, 32));
      r = fmaxf(r, __shfl_xor(r, 8, 32));
      float mn = fmaxf(mrun[v], r);
      float al = __expf(mrun[v] - mn);
      mrun[v] = mn;
      float p0 = __expf(s0 - mn), p1 = __expf(s1 - mn);
      float ps = p0 + p1;
      ps += __shfl_xor(ps, 1, 32);
      ps += __shfl_xor(ps, 2, 32);
      ps += __shfl_xor(ps, 4, 32);
      ps += __shfl_xor(ps, 8, 32);
      lrun[v] = lrun[v] * al + ps;
#pragma unroll
      for (int tt = 0; tt < 4; ++tt) o_[tt][v] *= al;
      int Mr = v + 8 * half;
      Pw[wave][Mr][l16]      = f2bf(p0);
      Pw[wave][Mr][16 + l16] = f2bf(p1);
    }
    __syncthreads();

    // O += P (16x32) . V (32x64)
    FragA pf;
#pragma unroll
    for (int p = 0; p < 8; ++p)
      pf.u[p] = *(const unsigned*)&Pw[wave][l16][kpair(half, p)];
#pragma unroll
    for (int tt = 0; tt < 4; ++tt) {
      FragA vf;
      int d = tt * 16 + l16;
#pragma unroll
      for (int p = 0; p < 8; ++p)
        vf.u[p] = *(const unsigned*)&Vt[d][kpair(half, p)];
      o_[tt] = __builtin_amdgcn_wmma_f32_16x16x32_bf16(
          false, pf.v, false, vf.v, (short)0, o_[tt], false, false);
    }
  }

  // epilogue: normalize and scatter-store bf16
#pragma unroll
  for (int v = 0; v < 8; ++v) {
    float inv = 1.0f / lrun[v];
    int Mr = v + 8 * half;
    unsigned short* rowp = O + (qrow0 + Mr) * 768 + h * HD + l16;
#pragma unroll
    for (int tt = 0; tt < 4; ++tt)
      rowp[tt * 16] = f2bf(o_[tt][v] * inv);
  }
}

// ---------------- host launch ----------------
extern "C" void kernel_launch(void* const* d_in, const int* in_sizes, int n_in,
                              void* d_out, int out_size, void* d_ws, size_t ws_size,
                              hipStream_t stream) {
  const float* x      = (const float*)d_in[0];   // [4,4096,768]
  const float* qkv_w  = (const float*)d_in[1];   // [2304,768]
  const float* proj_w = (const float*)d_in[2];   // [768,768]
  const float* proj_b = (const float*)d_in[3];   // [768]
  (void)in_sizes; (void)n_in; (void)out_size; (void)ws_size;

  char* ws = (char*)d_ws;
  size_t off = 0;
  auto take = [&](size_t bytes) { char* p = ws + off; off += (bytes + 255) & ~(size_t)255; return p; };

  unsigned short* xbf  = (unsigned short*)take((size_t)MQ  * 768 * 2);   // 25.2 MB
  unsigned short* wqbf = (unsigned short*)take((size_t)2304 * 768 * 2);  //  3.5 MB
  unsigned short* pwbf = (unsigned short*)take((size_t)768 * 768 * 2);   //  1.2 MB
  unsigned short* Qb   = (unsigned short*)take((size_t)MQ  * 768 * 2);   // 25.2 MB
  unsigned short* KVb  = (unsigned short*)take((size_t)MKV * 1536 * 2);  // 12.6 MB
  unsigned short* Ob   = xbf;  // x_bf dead after the QKV GEMMs: alias it

  dim3 blk(256);
  // 1) convert weights + activations to bf16
  cvt_f32_bf16_kernel<<<dim3((MQ * 768 / 4 + 255) / 256), blk, 0, stream>>>(x, xbf, MQ * 768);
  cvt_f32_bf16_kernel<<<dim3((2304 * 768 / 4 + 255) / 256), blk, 0, stream>>>(qkv_w, wqbf, 2304 * 768);
  cvt_f32_bf16_kernel<<<dim3((768 * 768 / 4 + 255) / 256), blk, 0, stream>>>(proj_w, pwbf, 768 * 768);

  // 2) Q projection: [16384,768] = X @ Wq^T   (block tile 128x256)
  gemm_bf16_kernel<<<dim3(MQ / 128, 768 / 256), blk, 0, stream>>>(
      xbf, 1, wqbf, Qb, nullptr, nullptr, 768);
  // 3) KV projection on subsampled rows: [4096,1536] = X[::4] @ Wkv^T
  gemm_bf16_kernel<<<dim3(MKV / 128, 1536 / 256), blk, 0, stream>>>(
      xbf, STRIDE, wqbf + (size_t)768 * 768, KVb, nullptr, nullptr, 1536);

  // 4) fused attention (flash-style, online softmax)
  attn_kernel<<<dim3(SLEN / 128, HNUM, NB), blk, 0, stream>>>(Qb, KVb, Ob);

  // 5) output projection + bias (f32 out)
  gemm_bf16_kernel<<<dim3(MQ / 128, 768 / 256), blk, 0, stream>>>(
      Ob, 1, pwbf, nullptr, (float*)d_out, proj_b, 768);
}
